// ScalerDecoder_27625229648410
// MI455X (gfx1250) — matrix operand order repaired and checked by reading, not compiled, not run on hardware
//
#include <hip/hip_runtime.h>

// CDNA5 / gfx1250: wave32, WMMA 16x16x32 bf16 with f32 accumulate.
typedef __attribute__((ext_vector_type(16))) __bf16 v16bf;
typedef __attribute__((ext_vector_type(8)))  __bf16 v8bf;
typedef __attribute__((ext_vector_type(8)))  float  v8f;

#define IN_F   256
#define HID    128
#define KT1    8          // IN_F / 32  (layer-1 k-tiles)
#define KT2    4          // HID  / 32  (layer-2 k-tiles)
#define APAD   8          // bf16 row pad -> +16B per row, bank-conflict-free
#define HPAD   8

__device__ __forceinline__ float silu_f(float x) {
    // x * sigmoid(x); v_rcp_f32 instead of IEEE divide chain
    return x * __builtin_amdgcn_rcpf(1.0f + __expf(-x));
}

// A-fragment (16x32 bf16) from a row-major bf16 LDS row, per CDNA5 WMMA layout:
// elems 0..7  = K[kt*32 + aOff + 0..7], elems 8..15 = K[kt*32 + 16 + aOff + 0..7]
__device__ __forceinline__ v16bf load_fragA(const __bf16* rowPtr, int kt, int aOff) {
    const v8bf lo = *(const v8bf*)(rowPtr + kt * 32 + aOff);
    const v8bf hi = *(const v8bf*)(rowPtr + kt * 32 + 16 + aOff);
    return __builtin_shufflevector(lo, hi,
        0, 1, 2, 3, 4, 5, 6, 7, 8, 9, 10, 11, 12, 13, 14, 15);
}

__global__ __launch_bounds__(256)
void scaler_decoder_fused(const float* __restrict__ scaler,      // [N, 256]
                          const int*   __restrict__ batch_index, // [N] sorted
                          const float* __restrict__ nW1,         // [256,128]
                          const float* __restrict__ nb1,         // [128]
                          const float* __restrict__ nW2,         // [128,128]
                          const float* __restrict__ nb2,         // [128]
                          const float* __restrict__ gW1,         // [256,128]
                          const float* __restrict__ gb1,         // [128]
                          const float* __restrict__ gW2,         // [128,128]
                          const float* __restrict__ gb2,         // [128]
                          const float* __restrict__ fW,          // [256]
                          const float* __restrict__ fb,          // [1]
                          float*       __restrict__ out,         // [G]
                          int N)
{
    __shared__ __attribute__((aligned(16))) __bf16 sA[2][16][IN_F + APAD]; // ping-pong node tiles
    __shared__ __attribute__((aligned(16))) __bf16 sH[16][HID  + HPAD];    // layer-1 activations
    __shared__ float sMean[IN_F];         // segment sum -> mean
    __shared__ float sNsPart[8][2][16];   // [wave(=ntile)][lane-half][col]
    __shared__ float sNs[HID];            // node_scaler (segment sum of node branch)
    __shared__ float sH1[HID];            // graph-branch hidden
    __shared__ float sG[HID];             // graph_scaler
    __shared__ float sRed[256];
    __shared__ int   sRange[2];

    const int g    = blockIdx.x;
    const int tid  = threadIdx.x;
    const int lane = tid & 31;
    const int wv   = tid >> 5;            // wave id == owned 16-col tile of HID

    // ---- graph node range via binary search on sorted batch_index ----
    if (tid == 0) {
        int lo = 0, hi = N;               // lower_bound(g)
        while (lo < hi) { int m = (lo + hi) >> 1; if (batch_index[m] < g) lo = m + 1; else hi = m; }
        sRange[0] = lo;
        hi = N;                           // lower_bound(g+1)
        int lo2 = lo;
        while (lo2 < hi) { int m = (lo2 + hi) >> 1; if (batch_index[m] < g + 1) lo2 = m + 1; else hi = m; }
        sRange[1] = lo2;
    }
    sMean[tid] = 0.0f;                    // zero-init for staged column-sum reduction
    __syncthreads();
    const int start  = sRange[0];
    const int end    = sRange[1];
    const int cnt    = end - start;
    const int ntiles = (cnt + 15) >> 4;

    // ---- build per-wave B fragments in registers (weights are L2-resident) ----
    const int nCol  = wv * 16 + (lane & 15);         // this lane's output column
    const int kOffB = (lane < 16) ? 0 : 8;           // B: lane-half K offset
    v16bf B1[KT1], B2[KT2];
#pragma unroll
    for (int kt = 0; kt < KT1; ++kt) {
        v16bf b;
#pragma unroll
        for (int e = 0; e < 16; ++e)
            b[e] = (__bf16)nW1[(size_t)(kt * 32 + kOffB + e + (e >= 8 ? 8 : 0)) * HID + nCol];
        B1[kt] = b;
    }
#pragma unroll
    for (int kt = 0; kt < KT2; ++kt) {
        v16bf b;
#pragma unroll
        for (int e = 0; e < 16; ++e)
            b[e] = (__bf16)nW2[(size_t)(kt * 32 + kOffB + e + (e >= 8 ? 8 : 0)) * HID + nCol];
        B2[kt] = b;
    }
    const float bias1 = nb1[nCol];
    const float bias2 = nb2[nCol];

    // ---- Phase 2: node-branch MLP via WMMA (double-buffered staging) ----
    const int rowA  = lane & 15;                     // A/C fragment row (low half)
    const int aOff  = (lane < 16) ? 0 : 8;           // A: lane-half K offset
    const int mHigh = (lane < 16) ? 0 : 8;           // C/D: lane-half row offset
    const int rStg  = tid >> 4;                      // staging row 0..15
    const int c0    = (tid & 15) * 16;               // staging col base

    float acc[8];                                    // node_scaler partials (per owned col)
#pragma unroll
    for (int r = 0; r < 8; ++r) acc[r] = 0.0f;
    float csum[16];                                  // scaler column-sum partials
#pragma unroll
    for (int i = 0; i < 16; ++i) csum[i] = 0.0f;

    // prologue: stage tile 0 into buffer 0 (and fold into column sums)
    if (ntiles > 0) {
        float st[16];
        const int node = start + rStg;
        if (node < end) {
            const float4* src = (const float4*)(scaler + (size_t)node * IN_F + c0);
#pragma unroll
            for (int q = 0; q < 4; ++q) {
                float4 v = src[q];
                st[4 * q + 0] = v.x; st[4 * q + 1] = v.y;
                st[4 * q + 2] = v.z; st[4 * q + 3] = v.w;
            }
        } else {
#pragma unroll
            for (int i = 0; i < 16; ++i) st[i] = 0.0f;
        }
#pragma unroll
        for (int i = 0; i < 16; ++i) csum[i] += st[i];
        v8bf lo8, hi8;
#pragma unroll
        for (int i = 0; i < 8; ++i) { lo8[i] = (__bf16)st[i]; hi8[i] = (__bf16)st[i + 8]; }
        *(v8bf*)&sA[0][rStg][c0]     = lo8;
        *(v8bf*)&sA[0][rStg][c0 + 8] = hi8;
    }
    __syncthreads();

    for (int ti = 0; ti < ntiles; ++ti) {
        const int cur = ti & 1;
        const int nxt = cur ^ 1;
        const int t0  = start + ti * 16;
        const bool haveNext = (ti + 1 < ntiles);     // uniform across block

        // (a) issue global loads for tile ti+1 now; consumed after compute
        float st[16];
        {
            const int node = t0 + 16 + rStg;
            if (haveNext && node < end) {
                const float4* src = (const float4*)(scaler + (size_t)node * IN_F + c0);
#pragma unroll
                for (int q = 0; q < 4; ++q) {
                    float4 v = src[q];
                    st[4 * q + 0] = v.x; st[4 * q + 1] = v.y;
                    st[4 * q + 2] = v.z; st[4 * q + 3] = v.w;
                }
            } else {
#pragma unroll
                for (int i = 0; i < 16; ++i) st[i] = 0.0f;
            }
        }

        // (b) layer 1: h[:, wv*16 .. wv*16+15] from sA[cur]
        // all 8 fragments preloaded (simultaneously live) -> one ds_load clause,
        // one dscnt wait, then 8 back-to-back WMMAs (C/D accumulator chaining)
        {
            const __bf16* aRow = &sA[cur][rowA][0];
            v16bf afr[KT1];
#pragma unroll
            for (int kt = 0; kt < KT1; ++kt)
                afr[kt] = load_fragA(aRow, kt, aOff);
            v8f c1 = {0.f, 0.f, 0.f, 0.f, 0.f, 0.f, 0.f, 0.f};
#pragma unroll
            for (int kt = 0; kt < KT1; ++kt)
                c1 = __builtin_amdgcn_wmma_f32_16x16x32_bf16(
                         false, afr[kt], false, B1[kt], (short)0, c1, false, false);
#pragma unroll
            for (int r = 0; r < 8; ++r) {
                const float h = silu_f(c1[r] + bias1);
                sH[r + mHigh][wv * 16 + (lane & 15)] = (__bf16)h;
            }
        }
        __syncthreads();                             // sH complete

        // (c) layer 2: out[:, wv*16 .. wv*16+15]; fold segment row-sum into acc
        {
            const __bf16* hRow = &sH[rowA][0];
            v16bf afr[KT2];
#pragma unroll
            for (int kt = 0; kt < KT2; ++kt)
                afr[kt] = load_fragA(hRow, kt, aOff);
            v8f c2 = {0.f, 0.f, 0.f, 0.f, 0.f, 0.f, 0.f, 0.f};
#pragma unroll
            for (int kt = 0; kt < KT2; ++kt)
                c2 = __builtin_amdgcn_wmma_f32_16x16x32_bf16(
                         false, afr[kt], false, B2[kt], (short)0, c2, false, false);
#pragma unroll
            for (int r = 0; r < 8; ++r) {
                const int node = t0 + r + mHigh;
                const float o  = silu_f(c2[r] + bias2);
                acc[r] += (node < end) ? o : 0.0f;   // mask padded rows
            }
        }

        // (d) convert staged regs -> sA[nxt]; fold into scaler column sums
        if (haveNext) {
#pragma unroll
            for (int i = 0; i < 16; ++i) csum[i] += st[i];   // zeros for invalid rows
            v8bf lo8, hi8;
#pragma unroll
            for (int i = 0; i < 8; ++i) { lo8[i] = (__bf16)st[i]; hi8[i] = (__bf16)st[i + 8]; }
            *(v8bf*)&sA[nxt][rStg][c0]     = lo8;
            *(v8bf*)&sA[nxt][rStg][c0 + 8] = hi8;
        }
        __syncthreads();                             // sA[nxt] ready; sH reads done
    }

    // ---- deterministic reduction of scaler column sums -> sMean (single writer/step) ----
    for (int rr = 0; rr < 16; ++rr) {
        if (rStg == rr) {
#pragma unroll
            for (int i = 0; i < 16; ++i) sMean[c0 + i] += csum[i];
        }
        __syncthreads();
    }
    {
        const float inv = 1.0f / (float)max(cnt, 1);
        sMean[tid] *= inv;                           // now the segment mean
    }

    // ---- reduce node_scaler: sum acc VGPRs, then combine lane halves via LDS ----
    {
        float s = 0.0f;
#pragma unroll
        for (int r = 0; r < 8; ++r) s += acc[r];
        sNsPart[wv][(lane < 16) ? 0 : 1][lane & 15] = s;
    }
    __syncthreads();
    if (tid < HID)
        sNs[tid] = sNsPart[tid >> 4][0][tid & 15] + sNsPart[tid >> 4][1][tid & 15];
    __syncthreads();

    // ---- Phase 3: graph-branch MLP (tiny, plain f32) ----
    if (tid < HID) {
        float a = gb1[tid];
        for (int k = 0; k < IN_F; ++k)
            a += sMean[k] * gW1[(size_t)k * HID + tid];
        sH1[tid] = silu_f(a);
    }
    __syncthreads();
    if (tid < HID) {
        float a = gb2[tid];
        for (int k = 0; k < HID; ++k)
            a += sH1[k] * gW2[(size_t)k * HID + tid];
        sG[tid] = silu_f(a);
    }
    __syncthreads();

    // ---- final: concat([graph_scaler, node_scaler]) @ fW + fb ----
    sRed[tid] = (tid < HID) ? sG[tid] * fW[tid] : sNs[tid - HID] * fW[tid];
    __syncthreads();
    for (int off = 128; off > 0; off >>= 1) {
        if (tid < off) sRed[tid] += sRed[tid + off];
        __syncthreads();
    }
    if (tid == 0) out[g] = sRed[0] + fb[0];
}

extern "C" void kernel_launch(void* const* d_in, const int* in_sizes, int n_in,
                              void* d_out, int out_size, void* d_ws, size_t ws_size,
                              hipStream_t stream) {
    // setup_inputs() order:
    // 0 pos, 1 mass_center, 2 scaler, 3 vector, 4 batch_index,
    // 5 nW1, 6 nb1, 7 nW2, 8 nb2, 9 gW1, 10 gb1, 11 gW2, 12 gb2, 13 fW, 14 fb
    const float* scaler      = (const float*)d_in[2];
    const int*   batch_index = (const int*)  d_in[4];
    const float* nW1 = (const float*)d_in[5];
    const float* nb1 = (const float*)d_in[6];
    const float* nW2 = (const float*)d_in[7];
    const float* nb2 = (const float*)d_in[8];
    const float* gW1 = (const float*)d_in[9];
    const float* gb1 = (const float*)d_in[10];
    const float* gW2 = (const float*)d_in[11];
    const float* gb2 = (const float*)d_in[12];
    const float* fW  = (const float*)d_in[13];
    const float* fb  = (const float*)d_in[14];
    float* out = (float*)d_out;

    const int N = in_sizes[4];        // number of nodes
    const int G = out_size;           // number of graphs (one block per graph)

    scaler_decoder_fused<<<G, 256, 0, stream>>>(
        scaler, batch_index, nW1, nb1, nW2, nb2,
        gW1, gb1, gW2, gb2, fW, fb, out, N);
}
